// Attention_63952063037623
// MI455X (gfx1250) — matrix operand order repaired and checked by reading, not compiled, not run on hardware
//
#include <hip/hip_runtime.h>
#include <hip/hip_bf16.h>

// ---------------------------------------------------------------------------
// Problem constants (B=32, N=577, C=1024, H=16, D=64)
// ---------------------------------------------------------------------------
#define BB   32
#define NN   577
#define CC   1024
#define HH   16
#define DD   64
#define NP   608          // N padded to multiple of 32 (19*32) for WMMA K loop
#define NKT  (NP/32)      // 19 K-steps in the PV GEMM
#define NTIL 37           // ceil(577/16)
#define MROWS (BB*NN)     // 18464 = 1154*16 (exact)

typedef __attribute__((ext_vector_type(8)))  __bf16 v8bf;
typedef __attribute__((ext_vector_type(16))) __bf16 v16bf;
typedef __attribute__((ext_vector_type(8)))  float  v8f;

// ---------------------------------------------------------------------------
// WMMA helpers (V_WMMA_F32_16X16X32_BF16, wave32)
//   A 16x32 bf16: M = lane%16, half = lane/16
//     elems j -> K = k0 + 16*half + j (contiguous 32B)
//   B 32x16 bf16: N = lane%16, elems j -> K = k0 + 16*half + j (contiguous 32B)
//   C/D f32: col = lane%16, row = r + 8*(lane/16)
// ---------------------------------------------------------------------------
__device__ __forceinline__ v16bf mk16(v8bf lo, v8bf hi) {
  return __builtin_shufflevector(lo, hi, 0,1,2,3,4,5,6,7,8,9,10,11,12,13,14,15);
}
__device__ __forceinline__ v16bf load_a_frag(const __bf16* row, int k0, int half) {
  v8bf lo = *reinterpret_cast<const v8bf*>(row + k0 + 16*half);
  v8bf hi = *reinterpret_cast<const v8bf*>(row + k0 + 16*half + 8);
  return mk16(lo, hi);
}
__device__ __forceinline__ v16bf load_b_frag(const __bf16* row, int k0, int half) {
  v8bf lo = *reinterpret_cast<const v8bf*>(row + k0 + 16*half);
  v8bf hi = *reinterpret_cast<const v8bf*>(row + k0 + 16*half + 8);
  return mk16(lo, hi);
}
__device__ __forceinline__ v8f wmma_bf16(v16bf a, v16bf b, v8f c) {
  return __builtin_amdgcn_wmma_f32_16x16x32_bf16(false, a, false, b, (short)0, c,
                                                 false, false);
}

// ---------------------------------------------------------------------------
// Async global->LDS staging (GLOBAL_LOAD_ASYNC_TO_LDS_B128, ASYNCcnt-tracked)
// Generic LDS pointer low 32 bits == byte offset within the wave's LDS alloc.
// ---------------------------------------------------------------------------
__device__ __forceinline__ void async_g2l_b128(void* lds_ptr, const void* gptr) {
  unsigned l = (unsigned)(size_t)lds_ptr;
  unsigned long long g = (unsigned long long)(size_t)gptr;
  asm volatile("global_load_async_to_lds_b128 %0, %1, off"
               :: "v"(l), "v"(g) : "memory");
}
__device__ __forceinline__ void wait_async_le2() {
  asm volatile("s_wait_asynccnt 0x2" ::: "memory");
}
__device__ __forceinline__ void wait_async_0() {
  asm volatile("s_wait_asynccnt 0x0" ::: "memory");
}

// ---------------------------------------------------------------------------
// K0: f32 -> bf16 cast
// ---------------------------------------------------------------------------
__global__ void cast_bf16_kernel(const float* __restrict__ in,
                                 __bf16* __restrict__ out, int n) {
  int i = blockIdx.x * blockDim.x + threadIdx.x;
  if (i < n) out[i] = (__bf16)in[i];
}

// ---------------------------------------------------------------------------
// K1: QKV GEMM.  qkv[i,o] = sum_c x[i,c] * w_qkv[o,c]
// grid (12, 1154), block 128 (4 waves). Wave = 16x64 output strip.
// ---------------------------------------------------------------------------
__global__ __launch_bounds__(128)
void qkv_gemm_kernel(const __bf16* __restrict__ x, const __bf16* __restrict__ w,
                     __bf16* __restrict__ q, __bf16* __restrict__ k,
                     __bf16* __restrict__ v) {
  const int lane = threadIdx.x & 31, wv = threadIdx.x >> 5;
  const int half = lane >> 4, ln = lane & 15;
  const int mbase = blockIdx.y * 16;
  const int n0w = blockIdx.x * 256 + wv * 64;

  const __bf16* arow = x + (size_t)(mbase + ln) * CC;
  const __bf16* brow[4];
#pragma unroll
  for (int t = 0; t < 4; ++t) brow[t] = w + (size_t)(n0w + t*16 + ln) * CC;

  v8f acc[4] = {};
  for (int k0 = 0; k0 < CC; k0 += 32) {
    v16bf a = load_a_frag(arow, k0, half);
#pragma unroll
    for (int t = 0; t < 4; ++t)
      acc[t] = wmma_bf16(a, load_b_frag(brow[t], k0, half), acc[t]);
  }

#pragma unroll
  for (int t = 0; t < 4; ++t) {
    int o = n0w + t*16 + ln;
    int which = o >> 10;           // 0=q 1=k 2=v
    int c = o & 1023;
    int h = c >> 6, d = c & 63;
    __bf16* dst = (which == 0) ? q : (which == 1) ? k : v;
#pragma unroll
    for (int r = 0; r < 8; ++r) {
      int i = mbase + r + 8*half;
      int b_ = i / NN, n_ = i - b_ * NN;
      dst[(((size_t)(b_*HH + h)) * NN + n_) * DD + d] = (__bf16)acc[t][r];
    }
  }
}

// ---------------------------------------------------------------------------
// K2: probe row-0 softmax per head, head-mean, top-k, prompt/patch additions.
// One block (256 threads) per batch. Tiny vs the GEMMs.
// ---------------------------------------------------------------------------
__global__ __launch_bounds__(256)
void probe_topk_apply_kernel(const __bf16* __restrict__ q,
                             __bf16* __restrict__ k, __bf16* __restrict__ v,
                             const float* __restrict__ prompt_k,
                             const float* __restrict__ prompt_v,
                             const float* __restrict__ patch_p_k,
                             const float* __restrict__ patch_p_v,
                             const int* __restrict__ kp) {
  __shared__ float logits[NN];
  __shared__ float cls[NN];
  __shared__ float red[256];
  __shared__ int   top[32];
  const int b = blockIdx.x, tid = threadIdx.x;

  for (int m = tid; m < NN; m += 256) cls[m] = 0.f;
  __syncthreads();

  for (int h = 0; h < HH; ++h) {
    const __bf16* qr = q + ((size_t)(b*HH + h) * NN) * DD;   // token 0
    for (int m = tid; m < NN; m += 256) {
      const __bf16* kr = k + ((size_t)(b*HH + h) * NN + m) * DD;
      float s = 0.f;
      for (int d = 0; d < DD; ++d) s += (float)qr[d] * (float)kr[d];
      logits[m] = s * 0.125f;
    }
    __syncthreads();
    float lm = -3.4e38f;
    for (int m = tid; m < NN; m += 256) lm = fmaxf(lm, logits[m]);
    red[tid] = lm; __syncthreads();
    for (int s = 128; s > 0; s >>= 1) {
      if (tid < s) red[tid] = fmaxf(red[tid], red[tid + s]);
      __syncthreads();
    }
    float mx = red[0]; __syncthreads();
    float ls = 0.f;
    for (int m = tid; m < NN; m += 256) ls += __expf(logits[m] - mx);
    red[tid] = ls; __syncthreads();
    for (int s = 128; s > 0; s >>= 1) {
      if (tid < s) red[tid] += red[tid + s];
      __syncthreads();
    }
    float inv = 1.0f / red[0]; __syncthreads();
    for (int m = tid; m < NN; m += 256)
      cls[m] += __expf(logits[m] - mx) * inv * (1.0f / (float)HH);
    __syncthreads();
  }

  int K = kp[0]; if (K > 32) K = 32; if (K < 0) K = 0;
  if (tid == 0) {
    for (int s = 0; s < K; ++s) {
      float best = -3.4e38f; int bi = 1;
      for (int m = 1; m < NN; ++m)
        if (cls[m] > best) { best = cls[m]; bi = m; }
      top[s] = bi; cls[bi] = -3.4e38f;
    }
  }
  __syncthreads();

  for (int j = tid; j < HH * DD; j += 256) {
    int h = j >> 6, d = j & 63;
    size_t base = ((size_t)(b*HH + h) * NN) * DD + d;
    k[base] = (__bf16)((float)k[base] + prompt_k[(b*HH + h) * DD + d]);
    v[base] = (__bf16)((float)v[base] + prompt_v[(b*HH + h) * DD + d]);
  }
  for (int s = 0; s < K; ++s) {
    int m = top[s];
    for (int j = tid; j < HH * DD; j += 256) {
      int h = j >> 6, d = j & 63;
      size_t a = ((size_t)(b*HH + h) * NN + m) * DD + d;
      k[a] = (__bf16)((float)k[a] + patch_p_k[h * DD + d]);
      v[a] = (__bf16)((float)v[a] + patch_p_v[h * DD + d]);
    }
  }
}

// ---------------------------------------------------------------------------
// K3: V (bh,n,d) -> Vt (bh,d,n) zero-padded to NP (PV GEMM B-operand is then
// contiguous along the WMMA K-dim).
// ---------------------------------------------------------------------------
__global__ void transpose_v_kernel(const __bf16* __restrict__ v,
                                   __bf16* __restrict__ vt) {
  size_t idx = (size_t)blockIdx.x * blockDim.x + threadIdx.x;
  const size_t total = (size_t)BB * HH * DD * NP;
  if (idx >= total) return;
  int tkn = (int)(idx % NP);
  size_t r = idx / NP;
  int d = (int)(r % DD);
  size_t bh = r / DD;
  vt[idx] = (tkn < NN) ? v[(bh * NN + tkn) * DD + d] : (__bf16)0.0f;
}

// ---------------------------------------------------------------------------
// K4: fused attention block. grid (37, H, B), block 128 (4 waves).
//   phase 0: kick off async V-chunk staging (overlaps phases 1-2)
//   phase 1: S(16xN) = Q_blk K^T * scale (WMMA)
//   phase 2: parallel row softmax (8 threads/row), exp folded into sum pass
//   phase 3: write attn f32 + bf16 probs; ctx_blk = P V (WMMA, B from
//            double-buffered async-staged LDS chunks, per-wave, no barriers)
// ---------------------------------------------------------------------------
__global__ __launch_bounds__(128)
void attn_kernel(const __bf16* __restrict__ q, const __bf16* __restrict__ k,
                 const __bf16* __restrict__ vt, float* __restrict__ attn,
                 __bf16* __restrict__ ctx) {
  __shared__ alignas(16) float  S[16][NP];         // 38,912 B
  __shared__ alignas(16) __bf16 P[16][NP];         // 19,456 B
  __shared__ alignas(16) __bf16 vbuf[4][2][16][32];//  8,192 B
  __shared__ float part[16][8];
  __shared__ float rmax[16], rinv[16];

  const int lane = threadIdx.x & 31, wv = threadIdx.x >> 5;
  const int half = lane >> 4, ln = lane & 15;
  const int b = blockIdx.z, h = blockIdx.y, bh = b * HH + h;
  const int m0 = blockIdx.x * 16;
  const int valid = (NN - m0) < 16 ? (NN - m0) : 16;

  // ---- phase 0: start async staging of V chunks 0 and 1 (per wave) --------
  const __bf16* vwave = vt + ((size_t)bh * DD + wv * 16) * NP;  // 16 d-rows
  {
    int rl = lane & 15, hp = lane >> 4;           // lane -> (row, 16B half)
#pragma unroll
    for (int c = 0; c < 2; ++c)
      async_g2l_b128(&vbuf[wv][0][rl][c*16 + hp*8],
                     vwave + (size_t)rl*NP + 0*32 + c*16 + hp*8);
#pragma unroll
    for (int c = 0; c < 2; ++c)
      async_g2l_b128(&vbuf[wv][1][rl][c*16 + hp*8],
                     vwave + (size_t)rl*NP + 1*32 + c*16 + hp*8);
  }

  // ---- phase 1: S = Q K^T * scale -----------------------------------------
  int ar = m0 + ln; if (ar > NN - 1) ar = NN - 1;     // clamp (masked later)
  const __bf16* arow = q + ((size_t)bh * NN + ar) * DD;
  for (int nt = wv; nt < NTIL; nt += 4) {
    int tc = nt * 16 + ln; if (tc > NN - 1) tc = NN - 1;
    const __bf16* brow = k + ((size_t)bh * NN + tc) * DD;
    v8f acc = {};
    acc = wmma_bf16(load_a_frag(arow, 0,  half), load_b_frag(brow, 0,  half), acc);
    acc = wmma_bf16(load_a_frag(arow, 32, half), load_b_frag(brow, 32, half), acc);
#pragma unroll
    for (int r = 0; r < 8; ++r)
      S[r + 8*half][nt*16 + ln] = acc[r] * 0.125f;
  }
  __syncthreads();

  // ---- phase 2: parallel row softmax (8 threads per row) ------------------
  {
    const int r = threadIdx.x >> 3, sub = threadIdx.x & 7;
    float lm = -3.4e38f;
    for (int n2 = sub; n2 < NN; n2 += 8) lm = fmaxf(lm, S[r][n2]);
    part[r][sub] = lm;
    __syncthreads();
    if (threadIdx.x < 16) {
      float mx = part[threadIdx.x][0];
#pragma unroll
      for (int j = 1; j < 8; ++j) mx = fmaxf(mx, part[threadIdx.x][j]);
      rmax[threadIdx.x] = mx;
    }
    __syncthreads();
    float mx = rmax[r], ls = 0.f;
    for (int n2 = sub; n2 < NN; n2 += 8) {
      float e = __expf(S[r][n2] - mx);
      S[r][n2] = e;                       // exp computed once, stored back
      ls += e;
    }
    part[r][sub] = ls;
    __syncthreads();
    if (threadIdx.x < 16) {
      float sm = 0.f;
#pragma unroll
      for (int j = 0; j < 8; ++j) sm += part[threadIdx.x][j];
      rinv[threadIdx.x] = 1.0f / sm;
    }
    __syncthreads();
  }

  // ---- write attn (f32 out) + P (bf16, zero padded) -----------------------
  for (int idx = threadIdx.x; idx < 16 * NP; idx += 128) {
    int m = idx / NP, n2 = idx % NP;
    float p = 0.f;
    if (m < valid && n2 < NN) {
      p = S[m][n2] * rinv[m];
      attn[((size_t)bh * NN + (m0 + m)) * NN + n2] = p;
    }
    P[m][n2] = (__bf16)p;
  }
  __syncthreads();

  // ---- phase 3: ctx_blk = P @ V (wave wv owns d-tile wv) ------------------
  {
    const __bf16* parow = &P[ln][0];
    const int rl = lane & 15, hp = lane >> 4;
    v8f acc = {};
    for (int kt = 0; kt < NKT; ++kt) {
      if (kt < NKT - 1) wait_async_le2();   // chunk kt landed, kt+1 in flight
      else              wait_async_0();     // last chunk: drain
      const int buf = kt & 1;
      v16bf a = load_a_frag(parow, kt * 32, half);
      const __bf16* vrow = &vbuf[wv][buf][ln][0];
      v8bf lo = *reinterpret_cast<const v8bf*>(vrow + 16*half);
      v8bf hi = *reinterpret_cast<const v8bf*>(vrow + 16*half + 8);
      acc = wmma_bf16(a, mk16(lo, hi), acc);
      if (kt + 2 < NKT) {                   // refill this buffer with kt+2
#pragma unroll
        for (int c = 0; c < 2; ++c)
          async_g2l_b128(&vbuf[wv][buf][rl][c*16 + hp*8],
                         vwave + (size_t)rl*NP + (kt + 2)*32 + c*16 + hp*8);
      }
    }
#pragma unroll
    for (int r = 0; r < 8; ++r) {
      int m = r + 8*half;
      if (m < valid)
        ctx[((size_t)(b * NN + m0 + m)) * CC + h*DD + wv*16 + ln] = (__bf16)acc[r];
    }
  }
}

// ---------------------------------------------------------------------------
// K5: projection GEMM. out[i,o] = sum_c ctx[i,c]*w_proj[o,c] + b_proj[o]
// grid (4, 1154), block 128.
// ---------------------------------------------------------------------------
__global__ __launch_bounds__(128)
void proj_gemm_kernel(const __bf16* __restrict__ ctx,
                      const __bf16* __restrict__ w,
                      const float* __restrict__ bias,
                      float* __restrict__ out) {
  const int lane = threadIdx.x & 31, wv = threadIdx.x >> 5;
  const int half = lane >> 4, ln = lane & 15;
  const int mbase = blockIdx.y * 16;
  const int n0w = blockIdx.x * 256 + wv * 64;

  const __bf16* arow = ctx + (size_t)(mbase + ln) * CC;
  const __bf16* brow[4];
#pragma unroll
  for (int t = 0; t < 4; ++t) brow[t] = w + (size_t)(n0w + t*16 + ln) * CC;

  v8f acc[4] = {};
  for (int k0 = 0; k0 < CC; k0 += 32) {
    v16bf a = load_a_frag(arow, k0, half);
#pragma unroll
    for (int t = 0; t < 4; ++t)
      acc[t] = wmma_bf16(a, load_b_frag(brow[t], k0, half), acc[t]);
  }
#pragma unroll
  for (int t = 0; t < 4; ++t) {
    int o = n0w + t*16 + ln;
    float bo = bias[o];
#pragma unroll
    for (int r = 0; r < 8; ++r) {
      int i = mbase + r + 8*half;
      out[(size_t)i * CC + o] = acc[t][r] + bo;
    }
  }
}

// ---------------------------------------------------------------------------
// Launch
// ---------------------------------------------------------------------------
extern "C" void kernel_launch(void* const* d_in, const int* in_sizes, int n_in,
                              void* d_out, int out_size, void* d_ws, size_t ws_size,
                              hipStream_t stream) {
  const float* x        = (const float*)d_in[0];
  const float* w_qkv    = (const float*)d_in[1];
  const float* w_proj   = (const float*)d_in[2];
  const float* b_proj   = (const float*)d_in[3];
  const float* patch_pk = (const float*)d_in[4];
  const float* patch_pv = (const float*)d_in[5];
  const float* prompt_k = (const float*)d_in[6];
  const float* prompt_v = (const float*)d_in[7];
  const int*   kp       = (const int*)d_in[8];

  float* out_proj = (float*)d_out;                         // (B,N,C)
  float* out_attn = out_proj + (size_t)BB * NN * CC;       // (B,H,N,N)

  const size_t XE  = (size_t)BB * NN * CC;   // 18,907,136
  const size_t WQE = (size_t)3 * CC * CC;    //  3,145,728
  const size_t WPE = (size_t)CC * CC;        //  1,048,576
  const size_t QE  = (size_t)BB * HH * NN * DD;
  const size_t VTE = (size_t)BB * HH * DD * NP;

  __bf16* xb   = (__bf16*)d_ws;
  __bf16* wqb  = xb  + XE;
  __bf16* wpb  = wqb + WQE;
  __bf16* qb   = wpb + WPE;
  __bf16* kb   = qb  + QE;
  __bf16* vb   = kb  + QE;
  __bf16* vtb  = vb  + QE;
  __bf16* ctxb = xb;   // alias: x no longer needed once QKV GEMM finished

  cast_bf16_kernel<<<(int)((XE  + 255) / 256), 256, 0, stream>>>(x, xb, (int)XE);
  cast_bf16_kernel<<<(int)((WQE + 255) / 256), 256, 0, stream>>>(w_qkv, wqb, (int)WQE);
  cast_bf16_kernel<<<(int)((WPE + 255) / 256), 256, 0, stream>>>(w_proj, wpb, (int)WPE);

  qkv_gemm_kernel<<<dim3(12, MROWS / 16), 128, 0, stream>>>(xb, wqb, qb, kb, vb);

  probe_topk_apply_kernel<<<BB, 256, 0, stream>>>(qb, kb, vb, prompt_k, prompt_v,
                                                  patch_pk, patch_pv, kp);

  transpose_v_kernel<<<(int)((VTE + 255) / 256), 256, 0, stream>>>(vb, vtb);

  attn_kernel<<<dim3(NTIL, HH, BB), 128, 0, stream>>>(qb, kb, vtb, out_attn, ctxb);

  proj_gemm_kernel<<<dim3(4, MROWS / 16), 128, 0, stream>>>(ctxb, wpb, b_proj,
                                                            out_proj);
}
